// RetrievalModule_38963943309333
// MI455X (gfx1250) — matrix operand order repaired and checked by reading, not compiled, not run on hardware
//
#include <hip/hip_runtime.h>
#include <hip/hip_bf16.h>
#include <cstdint>
#include <cstddef>

// ---------------------------------------------------------------------------
// Problem constants (from the reference)
// ---------------------------------------------------------------------------
#define DDIM   768
#define BQ     256
#define NDB    100000
#define TOPK   5
#define CH     64                         // db rows per chunk
#define NCHUNK ((NDB + CH - 1) / CH)      // 1563
#define NWG_MAIN 256
#define NEG_INF_F (-1.0e30f)

typedef _Float16 v8h  __attribute__((ext_vector_type(8)));
typedef _Float16 v16h __attribute__((ext_vector_type(16)));
typedef float    v8f  __attribute__((ext_vector_type(8)));
typedef float    v4f  __attribute__((ext_vector_type(4)));

static_assert(DDIM % 32 == 0, "D must be a multiple of WMMA K");

// ---------------------------------------------------------------------------
// Helpers
// ---------------------------------------------------------------------------
__device__ __forceinline__ v8f wmma_f16(v16h a, v16h b, v8f c) {
  // D = A(16x32 f16) * B(32x16 f16) + C(16x16 f32)
  return __builtin_amdgcn_wmma_f32_16x16x32_f16(
      /*neg_a=*/false, a, /*neg_b=*/false, b,
      /*c_mod=*/(short)0, c, /*reuse_a=*/false, /*reuse_b=*/false);
}

// A-fragment (this file's contiguous k order, B matches): per lane two contiguous 16B chunks,
// K = kb+{0..7} and K = kb+8+{0..7} where kb = (lane>=16) ? 16 : 0.
__device__ __forceinline__ v16h load_a_frag(const _Float16* __restrict__ p) {
  v8h lo = *(const v8h*)(p);
  v8h hi = *(const v8h*)(p + 8);
  v16h a;
#pragma unroll
  for (int i = 0; i < 8; ++i) { a[i] = lo[i]; a[i + 8] = hi[i]; }
  return a;
}

__device__ __forceinline__ void top5_insert(
    float v, int n,
    float& t0, float& t1, float& t2, float& t3, float& t4,
    int& i0, int& i1, int& i2, int& i3, int& i4) {
  if (v <= t4) return;
  if (v > t0)      { t4=t3;i4=i3; t3=t2;i3=i2; t2=t1;i2=i1; t1=t0;i1=i0; t0=v;i0=n; }
  else if (v > t1) { t4=t3;i4=i3; t3=t2;i3=i2; t2=t1;i2=i1; t1=v;i1=n; }
  else if (v > t2) { t4=t3;i4=i3; t3=t2;i3=i2; t2=v;i2=n; }
  else if (v > t3) { t4=t3;i4=i3; t3=v;i3=n; }
  else             { t4=v;i4=n; }
}

// ---------------------------------------------------------------------------
// Kernel 1: L2-normalize queries -> f16 (B x D)
// ---------------------------------------------------------------------------
__global__ __launch_bounds__(256)
void qnorm_kernel(const float* __restrict__ content, _Float16* __restrict__ qn) {
  const int q = blockIdx.x;
  const int t = threadIdx.x;
  const float* row = content + (size_t)q * DDIM;
  float ss = 0.f;
  for (int d = t; d < DDIM; d += 256) { float x = row[d]; ss += x * x; }
#pragma unroll
  for (int m = 16; m >= 1; m >>= 1) ss += __shfl_xor(ss, m, 32);
  __shared__ float red[8];
  __shared__ float rnS;
  if ((t & 31) == 0) red[t >> 5] = ss;
  __syncthreads();
  if (t == 0) {
    float s = 0.f;
#pragma unroll
    for (int i = 0; i < 8; ++i) s += red[i];
    rnS = 1.f / fmaxf(sqrtf(s), 1e-8f);   // matches torch-style eps clamp
  }
  __syncthreads();
  const float rn = rnS;
  for (int d = t; d < DDIM; d += 256)
    qn[(size_t)q * DDIM + d] = (_Float16)(row[d] * rn);
}

// ---------------------------------------------------------------------------
// Kernel 2: convert W1 / W2 to f16 (row-major, rows contiguous over K)
// ---------------------------------------------------------------------------
__global__ __launch_bounds__(256)
void cvt_w_kernel(const float* __restrict__ W1, const float* __restrict__ W2,
                  _Float16* __restrict__ w1h, _Float16* __restrict__ w2h) {
  const int n1 = DDIM * 2 * DDIM;   // 1179648
  const int n2 = DDIM * DDIM;       // 589824
  for (int i = blockIdx.x * blockDim.x + threadIdx.x; i < n1 + n2;
       i += gridDim.x * blockDim.x) {
    if (i < n1) w1h[i] = (_Float16)W1[i];
    else        w2h[i - n1] = (_Float16)W2[i - n1];
  }
}

// ---------------------------------------------------------------------------
// Kernel 3: fused  (cosine-sim GEMM via WMMA) + speaker mask + running top-5
//   Block = 256 threads = 8 waves. Per chunk of 64 db rows:
//   - stage fp32 rows -> f16 in LDS, accumulate per-row sum-of-squares
//   - each wave: 2 M-tiles (32 queries) x 4 N-tiles, WMMA f16->f32 accum
//   - scale by rsqrt(ss), mask by speaker, spill to LDS, per-thread top-5
// ---------------------------------------------------------------------------
__global__ __launch_bounds__(256)
void sims_topk_kernel(const float* __restrict__ db, const int* __restrict__ spk_ids,
                      const int* __restrict__ tgt, const _Float16* __restrict__ qn,
                      float* __restrict__ pvals, int* __restrict__ pidx) {
  __shared__ _Float16 sB[CH * 32];        // 4 KB staged f16 B-tile (row-major 64x32)
  __shared__ float    sRn[CH];            // per-row rsqrt(sum x^2)
  __shared__ int      sSpk[CH];           // per-row speaker id
  __shared__ float    sSims[128 * 65];    // 33 KB padded sims (half the queries/round)

  const int tid  = threadIdx.x;
  const int lane = tid & 31;
  const int w    = tid >> 5;              // wave 0..7
  const int hsel = (lane >> 4) & 1;       // lane half within the wave

  // Persistent per-thread top-5 (thread tid owns query tid).
  float tv0 = -3e38f, tv1 = -3e38f, tv2 = -3e38f, tv3 = -3e38f, tv4 = -3e38f;
  int   ti0 = 0, ti1 = 0, ti2 = 0, ti3 = 0, ti4 = 0;

  // Target speaker id for every (a-tile, accumulator row) this lane touches.
  int tgtv[2][8];
#pragma unroll
  for (int a = 0; a < 2; ++a)
#pragma unroll
    for (int r = 0; r < 8; ++r)
      tgtv[a][r] = tgt[w * 32 + a * 16 + r + hsel * 8];

  const int rowL = tid >> 2;              // staging row 0..63 (4 threads/row)
  const int cj   = (tid & 3) * 8;         // staging column offset (8 floats)

  for (int chunk = blockIdx.x; chunk < NCHUNK; chunk += gridDim.x) {
    const int nbase = chunk * CH;
    if (tid < CH) {
      const int n = nbase + tid;
      sSpk[tid] = (n < NDB) ? spk_ids[n] : -1;
    }

    v8f acc[2][4];
#pragma unroll
    for (int a = 0; a < 2; ++a)
#pragma unroll
      for (int j = 0; j < 4; ++j)
#pragma unroll
        for (int e = 0; e < 8; ++e) acc[a][j][e] = 0.f;

    float ss = 0.f;
    const int  nrow  = nbase + rowL;
    const bool rowOK = (nrow < NDB);
    const float* rp  = db + (size_t)nrow * DDIM + cj;

    for (int k0 = 0; k0 < DDIM; k0 += 32) {
      __syncthreads();                    // previous sB reads complete
      // ---- stage 64x32 fp32 tile -> f16 LDS, accumulate sum-of-squares ----
      v4f x0 = {0.f, 0.f, 0.f, 0.f}, x1 = {0.f, 0.f, 0.f, 0.f};
      if (rowOK) {
        x0 = *(const v4f*)(rp + k0);
        x1 = *(const v4f*)(rp + k0 + 4);
        if (k0 + 32 < DDIM) __builtin_prefetch(rp + k0 + 32, 0, 3);
      }
      ss += x0.x * x0.x + x0.y * x0.y + x0.z * x0.z + x0.w * x0.w +
            x1.x * x1.x + x1.y * x1.y + x1.z * x1.z + x1.w * x1.w;
      v8h hx;
      hx[0] = (_Float16)x0.x; hx[1] = (_Float16)x0.y;
      hx[2] = (_Float16)x0.z; hx[3] = (_Float16)x0.w;
      hx[4] = (_Float16)x1.x; hx[5] = (_Float16)x1.y;
      hx[6] = (_Float16)x1.z; hx[7] = (_Float16)x1.w;
      *(v8h*)(sB + rowL * 32 + cj) = hx;
      __syncthreads();

      // ---- WMMA: 2 A-tiles x 4 B-tiles per wave ----
      v16h afr[2];
#pragma unroll
      for (int a = 0; a < 2; ++a) {
        const _Float16* ap =
            qn + (size_t)(w * 32 + a * 16 + (lane & 15)) * DDIM + k0 + hsel * 16;
        afr[a] = load_a_frag(ap);
      }
#pragma unroll
      for (int j = 0; j < 4; ++j) {
        const v16h bfr = *(const v16h*)(sB + (j * 16 + (lane & 15)) * 32 + hsel * 16);
#pragma unroll
        for (int a = 0; a < 2; ++a) acc[a][j] = wmma_f16(afr[a], bfr, acc[a][j]);
      }
    }

    // ---- finalize per-row rsqrt(sum x^2) ----
    ss += __shfl_xor(ss, 1, 32);
    ss += __shfl_xor(ss, 2, 32);
    if ((tid & 3) == 0) sRn[rowL] = rsqrtf(fmaxf(ss, 1e-16f));
    __syncthreads();

    // ---- normalize, mask, spill to LDS, update running top-5 (2 rounds) ----
    for (int round = 0; round < 2; ++round) {
      if ((w >> 2) == round) {
        const int wl = w & 3;
#pragma unroll
        for (int a = 0; a < 2; ++a)
#pragma unroll
          for (int j = 0; j < 4; ++j) {
            const int   nl = j * 16 + (lane & 15);
            const float rn = sRn[nl];
            const int   sp = sSpk[nl];
#pragma unroll
            for (int r = 0; r < 8; ++r) {
              const int qloc = wl * 32 + a * 16 + r + hsel * 8;  // 0..127
              const float v  = (sp == tgtv[a][r]) ? acc[a][j][r] * rn : NEG_INF_F;
              sSims[qloc * 65 + nl] = v;
            }
          }
      }
      __syncthreads();
      if ((tid >> 7) == round) {
        const int qloc = tid & 127;
        const float* sp2 = &sSims[qloc * 65];
        for (int i = 0; i < CH; ++i)
          top5_insert(sp2[i], nbase + i, tv0, tv1, tv2, tv3, tv4,
                      ti0, ti1, ti2, ti3, ti4);
      }
      __syncthreads();
    }
  }

  // ---- write this WG's per-query partial top-5 ----
  const size_t base = ((size_t)blockIdx.x * BQ + tid) * TOPK;
  pvals[base + 0] = tv0; pvals[base + 1] = tv1; pvals[base + 2] = tv2;
  pvals[base + 3] = tv3; pvals[base + 4] = tv4;
  pidx[base + 0] = ti0; pidx[base + 1] = ti1; pidx[base + 2] = ti2;
  pidx[base + 3] = ti3; pidx[base + 4] = ti4;
}

// ---------------------------------------------------------------------------
// Kernel 4: merge NWG_MAIN partial top-5 lists per query -> final indices
// ---------------------------------------------------------------------------
__global__ __launch_bounds__(128)
void topk_reduce_kernel(const float* __restrict__ pvals, const int* __restrict__ pidx,
                        int* __restrict__ topidx) {
  const int q = blockIdx.x;
  const int t = threadIdx.x;
  __shared__ float sv[128 * TOPK];
  __shared__ int   si[128 * TOPK];
  float t0 = -3e38f, t1 = -3e38f, t2 = -3e38f, t3 = -3e38f, t4 = -3e38f;
  int   i0 = 0, i1 = 0, i2 = 0, i3 = 0, i4 = 0;
  for (int g = t; g < NWG_MAIN; g += 128) {
    const size_t base = ((size_t)g * BQ + q) * TOPK;
#pragma unroll
    for (int i = 0; i < TOPK; ++i)
      top5_insert(pvals[base + i], pidx[base + i], t0, t1, t2, t3, t4,
                  i0, i1, i2, i3, i4);
  }
  sv[t * TOPK + 0] = t0; sv[t * TOPK + 1] = t1; sv[t * TOPK + 2] = t2;
  sv[t * TOPK + 3] = t3; sv[t * TOPK + 4] = t4;
  si[t * TOPK + 0] = i0; si[t * TOPK + 1] = i1; si[t * TOPK + 2] = i2;
  si[t * TOPK + 3] = i3; si[t * TOPK + 4] = i4;
  __syncthreads();
  if (t == 0) {
    float m0 = -3e38f, m1 = -3e38f, m2 = -3e38f, m3 = -3e38f, m4 = -3e38f;
    int   j0 = 0, j1 = 0, j2 = 0, j3 = 0, j4 = 0;
    for (int i = 0; i < 128 * TOPK; ++i)
      top5_insert(sv[i], si[i], m0, m1, m2, m3, m4, j0, j1, j2, j3, j4);
    topidx[q * TOPK + 0] = j0; topidx[q * TOPK + 1] = j1;
    topidx[q * TOPK + 2] = j2; topidx[q * TOPK + 3] = j3;
    topidx[q * TOPK + 4] = j4;
  }
}

// ---------------------------------------------------------------------------
// Kernel 5: gather retrieved = mean of top-5 rows; build combined (f16, Bx1536)
// ---------------------------------------------------------------------------
__global__ __launch_bounds__(256)
void gather_kernel(const float* __restrict__ content, const float* __restrict__ db,
                   const int* __restrict__ topidx, _Float16* __restrict__ comb) {
  const int q = blockIdx.x;
  const int t = threadIdx.x;
  __shared__ int idx[TOPK];
  if (t < TOPK) idx[t] = topidx[q * TOPK + t];
  __syncthreads();
  for (int d = t; d < DDIM; d += 256) {
    comb[(size_t)q * (2 * DDIM) + d] = (_Float16)content[(size_t)q * DDIM + d];
    float a = 0.f;
#pragma unroll
    for (int i = 0; i < TOPK; ++i) a += db[(size_t)idx[i] * DDIM + d];
    comb[(size_t)q * (2 * DDIM) + DDIM + d] = (_Float16)(a * 0.2f);
  }
}

// ---------------------------------------------------------------------------
// Kernel 6/7: WMMA MLP layer. One wave per 16x16 output tile.
//   h = silu(A @ W^T + b)  -> f16,  or  out = A @ W^T + b -> f32
// ---------------------------------------------------------------------------
template <int KDIM, bool SILU>
__global__ __launch_bounds__(256)
void mlp_kernel(const _Float16* __restrict__ Ain, const _Float16* __restrict__ Wh,
                const float* __restrict__ bias, _Float16* __restrict__ out16,
                float* __restrict__ out32, int tilesN) {
  const int lane = threadIdx.x & 31;
  const int w    = threadIdx.x >> 5;
  const int hsel = (lane >> 4) & 1;
  const int tileId = blockIdx.x * 8 + w;
  const int tm = tileId / tilesN;
  const int tn = tileId % tilesN;
  const int m = tm * 16 + (lane & 15);
  const int n = tn * 16 + (lane & 15);

  v8f acc;
#pragma unroll
  for (int e = 0; e < 8; ++e) acc[e] = 0.f;

#pragma unroll 4
  for (int k0 = 0; k0 < KDIM; k0 += 32) {
    const _Float16* ap = Ain + (size_t)m * KDIM + k0 + hsel * 16;
    const v16h afr = load_a_frag(ap);
    const v16h bfr = *(const v16h*)(Wh + (size_t)n * KDIM + k0 + hsel * 16);
    acc = wmma_f16(afr, bfr, acc);
  }

  const float bn = bias[n];
  const int nCols = tilesN * 16;
#pragma unroll
  for (int r = 0; r < 8; ++r) {
    const int mr = tm * 16 + r + hsel * 8;
    const float x = acc[r] + bn;
    if (SILU) {
      const float s = x / (1.f + __expf(-x));
      out16[(size_t)mr * nCols + n] = (_Float16)s;
    } else {
      out32[(size_t)mr * nCols + n] = x;
    }
  }
}

// ---------------------------------------------------------------------------
// Launch
// ---------------------------------------------------------------------------
extern "C" void kernel_launch(void* const* d_in, const int* in_sizes, int n_in,
                              void* d_out, int out_size, void* d_ws, size_t ws_size,
                              hipStream_t stream) {
  (void)in_sizes; (void)n_in; (void)out_size; (void)ws_size;

  const float* content = (const float*)d_in[0];
  const int*   tgt     = (const int*)d_in[1];
  const float* db      = (const float*)d_in[2];
  const int*   spk     = (const int*)d_in[3];
  const float* W1      = (const float*)d_in[4];
  const float* b1      = (const float*)d_in[5];
  const float* W2      = (const float*)d_in[6];
  const float* b2      = (const float*)d_in[7];
  float* out = (float*)d_out;

  char* ws = (char*)d_ws;
  size_t off = 0;
  auto alloc = [&](size_t bytes) -> void* {
    void* p = ws + off;
    off = (off + bytes + 255) & ~(size_t)255;
    return p;
  };
  _Float16* qn   = (_Float16*)alloc((size_t)BQ * DDIM * 2);
  _Float16* w1h  = (_Float16*)alloc((size_t)DDIM * 2 * DDIM * 2);
  _Float16* w2h  = (_Float16*)alloc((size_t)DDIM * DDIM * 2);
  _Float16* comb = (_Float16*)alloc((size_t)BQ * 2 * DDIM * 2);
  _Float16* h16  = (_Float16*)alloc((size_t)BQ * DDIM * 2);
  float* pvals   = (float*)alloc((size_t)NWG_MAIN * BQ * TOPK * 4);
  int*   pidx    = (int*)alloc((size_t)NWG_MAIN * BQ * TOPK * 4);
  int*   topidx  = (int*)alloc((size_t)BQ * TOPK * 4);

  qnorm_kernel<<<BQ, 256, 0, stream>>>(content, qn);
  cvt_w_kernel<<<864, 256, 0, stream>>>(W1, W2, w1h, w2h);
  sims_topk_kernel<<<NWG_MAIN, 256, 0, stream>>>(db, spk, tgt, qn, pvals, pidx);
  topk_reduce_kernel<<<BQ, 128, 0, stream>>>(pvals, pidx, topidx);
  gather_kernel<<<BQ, 256, 0, stream>>>(content, db, topidx, comb);
  mlp_kernel<2 * DDIM, true><<<(BQ / 16) * 48 / 8, 256, 0, stream>>>(
      comb, w1h, b1, h16, nullptr, 48);
  mlp_kernel<DDIM, false><<<(BQ / 16) * 48 / 8, 256, 0, stream>>>(
      h16, w2h, b2, nullptr, out, 48);
}
